// R2D2AtariNetwork_69432441307617
// MI455X (gfx1250) — matrix operand order repaired and checked by reading, not compiled, not run on hardware
//
#include <hip/hip_runtime.h>

// ---------------------------------------------------------------------------
// R2D2 Atari network for MI455X (gfx1250, wave32, WMMA + TDM).
// All GEMM-like work uses v_wmma_f32_16x16x32_f16 (f16 in, f32 accumulate).
// Conv activation tiles are staged to LDS via the Tensor Data Mover.
// ---------------------------------------------------------------------------

typedef _Float16 half_t;
typedef __attribute__((ext_vector_type(16))) _Float16 v16h;
typedef __attribute__((ext_vector_type(8)))  float    v8f;
typedef __attribute__((ext_vector_type(4)))  unsigned int u32x4;
typedef __attribute__((ext_vector_type(8)))  int          i32x8;
typedef __attribute__((ext_vector_type(4)))  int          i32x4;

union ABu { v16h v; unsigned int u[8]; };

__device__ inline v8f zero_v8f() {
  v8f z;
#pragma unroll
  for (int i = 0; i < 8; ++i) z[i] = 0.0f;
  return z;
}

// A fragment (16x32 f16, M x K), source tile row-major [16][ld].
__device__ inline v16h load_frag_a(const half_t* p, int ld, int lane) {
  ABu f;
  int m = lane & 15, hf = lane >> 4;
#pragma unroll
  for (int v = 0; v < 8; ++v) {
    int k = ((v < 4) ? 0 : 16) + hf * 8 + (v & 3) * 2;
    f.u[v] = *(const unsigned int*)(p + m * ld + k);
  }
  return f.v;
}

// B fragment (32x16 f16, K x N), source tile stored N-major [16][ld].
__device__ inline v16h load_frag_b(const half_t* p, int ld, int lane) {
  ABu f;
  int n = lane & 15, hf = lane >> 4;
#pragma unroll
  for (int v = 0; v < 8; ++v) {
    int k = ((v < 4) ? 0 : 16) + hf * 8 + (v & 3) * 2;
    f.u[v] = *(const unsigned int*)(p + n * ld + k);
  }
  return f.v;
}

__device__ inline float sigm(float x) { return 1.0f / (1.0f + __expf(-x)); }

// ---------------------------------------------------------------------------
// Tensor Data Mover: 1D contiguous global -> LDS copy (data_size = 8 bytes).
// D# per CDNA5 ISA 8.3/8.4. Issued by one lane (EXEC ignored for TDM ops),
// completion via s_wait_tensorcnt. nu8 = bytes/8 (must fit tile_dim0 u16).
// This toolchain exposes the 6-arg builtin:
//   (uint32x4 g0, int32x8 g1, int32x4 g2, int32x4 g3, int32x8 extra, i32 cpol)
// Groups beyond 0/1 are unused for a <=2D single-row tile -> zero-filled.
// ---------------------------------------------------------------------------
#if __has_builtin(__builtin_amdgcn_tensor_load_to_lds)
#define HAVE_TDM 1
__device__ inline void tdm_load_1d(unsigned long long gaddr, unsigned int lds_off,
                                   unsigned int nu8) {
  u32x4 g0;
  g0[0] = 1u;                                   // count=1, user descriptor
  g0[1] = lds_off;                              // lds_addr (bytes)
  g0[2] = (unsigned int)(gaddr & 0xffffffffu);  // global_addr[31:0]
  g0[3] = (unsigned int)((gaddr >> 32) & 0x01ffffffu) | (2u << 30);  // [56:32]|type=2
  i32x8 g1;
  g1[0] = (int)(3u << 16);                      // data_size=8B; no multicast/pad
  g1[1] = (int)((nu8 & 0xffffu) << 16);         // tensor_dim0[15:0] @bits63:48
  g1[2] = (int)((nu8 >> 16) | (1u << 16));      // tensor_dim0[31:16]; tensor_dim1=1
  g1[3] = (int)((nu8 & 0xffffu) << 16);         // tensor_dim1 hi=0; tile_dim0
  g1[4] = 1;                                    // tile_dim1=1, tile_dim2=0
  g1[5] = (int)nu8;                             // tensor_dim0_stride[31:0]
  g1[6] = 0;                                    // stride hi, dim1_stride lo
  g1[7] = 0;
  i32x4 gz4;
  gz4[0] = 0; gz4[1] = 0; gz4[2] = 0; gz4[3] = 0;
  i32x8 gz8;
#pragma unroll
  for (int i = 0; i < 8; ++i) gz8[i] = 0;
  __builtin_amdgcn_tensor_load_to_lds(g0, g1, gz4, gz4, gz8, 0);
}
#else
#define HAVE_TDM 0
#endif

// Stage nu8*8 bytes from global `src` into shared `dst`, whole-block helper.
__device__ inline void stage_tile(half_t* dst, const half_t* src, int nu8, int tid) {
#if HAVE_TDM
  if (tid == 0) {
    tdm_load_1d((unsigned long long)(uintptr_t)src,
                (unsigned int)(uintptr_t)dst,  // low 32 bits = LDS byte address
                (unsigned int)nu8);
  }
  __builtin_amdgcn_s_wait_tensorcnt((short)0);
  __syncthreads();
#else
  int n16 = nu8 >> 1;  // uint4 count
  for (int i = tid; i < n16; i += 256)
    ((uint4*)dst)[i] = ((const uint4*)src)[i];
  __syncthreads();
#endif
}

// ---------------------------------------------------------------------------
// Weight repack: everything to f16 in WMMA-friendly [M][K] layouts.
// ---------------------------------------------------------------------------
__global__ __launch_bounds__(256) void pack_weights(
    const float* c1w, const float* c2w, const float* c3w, const float* fcw,
    const float* wih, const float* whh, const float* bih, const float* bhh,
    const float* aw1, const float* aw2, const float* vw1, const float* vw2,
    half_t* W1p, half_t* W2p, half_t* W3p, half_t* FCp, half_t* WLp,
    half_t* A1p, half_t* A2p, half_t* V1p, half_t* V2p, float* bsum) {
  const long long N1 = 2048, N2 = 32768, N3 = 36864, NF = 512LL * 3136,
                  NL = 2048LL * 1088, NA1 = 262144, NA2 = 32768, NV1 = 262144,
                  NV2 = 32768, NB = 2048;
  long long total = N1 + N2 + N3 + NF + NL + NA1 + NA2 + NV1 + NV2 + NB;
  for (long long idx = (long long)blockIdx.x * 256 + threadIdx.x; idx < total;
       idx += (long long)gridDim.x * 256) {
    long long i = idx;
    if (i < N1) { W1p[i] = (half_t)c1w[i]; continue; }
    i -= N1;
    if (i < N2) {  // conv2: K = (ky*4+kx)*32 + c  (NHWC im2col)
      int ii = (int)i, o = ii / 512, k = ii % 512;
      int ky = k >> 7, kx = (k >> 5) & 3, c = k & 31;
      W2p[ii] = (half_t)c2w[((o * 32 + c) * 4 + ky) * 4 + kx];
      continue;
    }
    i -= N2;
    if (i < N3) {  // conv3: K = (ky*3+kx)*64 + c
      int ii = (int)i, o = ii / 576, k = ii % 576;
      int p = k >> 6, c = k & 63, ky = p / 3, kx = p % 3;
      W3p[ii] = (half_t)c3w[((o * 64 + c) * 3 + ky) * 3 + kx];
      continue;
    }
    i -= N3;
    if (i < NF) {  // fc: reorder K from c*49+s (NCHW flatten) to s*64+c (NHWC act3)
      int m = (int)(i / 3136), k = (int)(i % 3136);
      int s = k >> 6, c = k & 63;
      FCp[i] = (half_t)fcw[m * 3136 + c * 49 + s];
      continue;
    }
    i -= NF;
    if (i < NL) {  // LSTM: [2048][1088] = [W_ih(531) pad13 | W_hh(512) pad32]
      int m = (int)(i / 1088), k = (int)(i % 1088);
      float v = 0.0f;
      if (k < 531) v = wih[m * 531 + k];
      else if (k >= 544 && k < 1056) v = whh[m * 512 + (k - 544)];
      WLp[i] = (half_t)v;
      continue;
    }
    i -= NL;
    if (i < NA1) { A1p[i] = (half_t)aw1[i]; continue; }
    i -= NA1;
    if (i < NA2) {  // adv_w2 padded 18 -> 64 rows
      int m = (int)(i / 512), k = (int)(i % 512);
      A2p[i] = (half_t)((m < 18) ? aw2[m * 512 + k] : 0.0f);
      continue;
    }
    i -= NA2;
    if (i < NV1) { V1p[i] = (half_t)vw1[i]; continue; }
    i -= NV1;
    if (i < NV2) {  // val_w2 padded 1 -> 64 rows
      int m = (int)(i / 512), k = (int)(i % 512);
      V2p[i] = (half_t)((m < 1) ? vw2[k] : 0.0f);
      continue;
    }
    i -= NV2;
    bsum[i] = bih[i] + bhh[i];
  }
}

__global__ __launch_bounds__(256) void init_states(const float* h0, const float* c0,
                                                   half_t* hstate_h, float* hstate_f,
                                                   float* cstate_f) {
  int i = blockIdx.x * 256 + threadIdx.x;
  if (i < 32768) {
    hstate_h[i] = (half_t)h0[i];
    hstate_f[i] = h0[i];
    cstate_f[i] = c0[i];
  }
}

// ---------------------------------------------------------------------------
// conv1: 1x84x84 -> 32x20x20, 8x8 stride4, direct VALU (K=64, tiny FLOP share).
// Output NHWC f16: act1[img][oy*20+ox][c].
// ---------------------------------------------------------------------------
__global__ __launch_bounds__(256) void conv1_direct(const int* o, const float* w,
                                                    const float* bias, half_t* act1) {
  __shared__ float img[7056];
  __shared__ float wsm[2048];
  __shared__ float bsm[32];
  int imgid = blockIdx.x, tid = threadIdx.x;
  const int* src = o + (size_t)imgid * 7056;
  for (int i = tid; i < 7056; i += 256) img[i] = (float)src[i] * (1.0f / 255.0f);
  for (int i = tid; i < 2048; i += 256) wsm[i] = w[i];
  if (tid < 32) bsm[tid] = bias[tid];
  __syncthreads();
  for (int idx = tid; idx < 12800; idx += 256) {
    int c = idx & 31, p = idx >> 5;
    int oy = p / 20, ox = p % 20;
    float s = bsm[c];
    const float* wp = wsm + c * 64;
    const float* ip = img + (oy * 4) * 84 + ox * 4;
#pragma unroll
    for (int ky = 0; ky < 8; ++ky)
#pragma unroll
      for (int kx = 0; kx < 8; ++kx)
        s = fmaf(ip[ky * 84 + kx], wp[ky * 8 + kx], s);
    act1[(size_t)imgid * 12800 + (size_t)p * 32 + c] = (half_t)fmaxf(s, 0.0f);
  }
}

// ---------------------------------------------------------------------------
// conv2: implicit GEMM. One WG per frame. M=64 outch, N=81 pix (pad 96), K=512.
// 20x20x32 NHWC tile staged in LDS via TDM; B fragments im2col-gathered.
// Padded columns (npix>=81) use a clamped in-bounds address; their garbage
// only affects output columns that are never stored (D col n <- B col n).
// ---------------------------------------------------------------------------
__global__ __launch_bounds__(256) void conv2_wmma(const half_t* act1, const half_t* W2p,
                                                  const float* b2, half_t* act2) {
  __shared__ half_t a1[20 * 20 * 32];  // 25.6 KB
  int img = blockIdx.x, tid = threadIdx.x, lane = tid & 31, wv = tid >> 5;
  stage_tile(a1, act1 + (size_t)img * 12800, 3200, tid);
  int hf = lane >> 4;
  for (int rr = 0; rr < 3; ++rr) {
    int tile = wv + rr * 8;  // 24 tiles = 4 m-tiles x 6 n-tiles
    int mt = tile & 3, nt = tile >> 2;
    int m = mt * 16 + (lane & 15);
    int npix = nt * 16 + (lane & 15);
    bool valid = npix < 81;
    int npc = valid ? npix : 80;  // clamp address, no exec toggling in K loop
    int oy = npc / 9, ox = npc % 9;
    v8f acc = zero_v8f();
    for (int kk = 0; kk < 512; kk += 32) {
      ABu fa, fb;
#pragma unroll
      for (int v = 0; v < 8; ++v) {
        int ka = kk + ((v < 4) ? 0 : 16) + hf * 8 + (v & 3) * 2;
        fa.u[v] = *(const unsigned int*)(W2p + m * 512 + ka);
        int kb = kk + ((v < 4) ? 0 : 16) + hf * 8 + (v & 3) * 2;
        int ky = kb >> 7, kx = (kb >> 5) & 3, c = kb & 31;
        int offb = ((oy * 2 + ky) * 20 + (ox * 2 + kx)) * 32 + c;
        fb.u[v] = *(const unsigned int*)(a1 + offb);
      }
      acc = __builtin_amdgcn_wmma_f32_16x16x32_f16(false, fa.v, false, fb.v,
                                                   (short)0, acc, false, false);
    }
#pragma unroll
    for (int v = 0; v < 8; ++v) {
      int mo = mt * 16 + (lane >> 4) * 8 + v;
      float val = fmaxf(acc[v] + b2[mo], 0.0f);
      if (valid) act2[((size_t)img * 81 + npix) * 64 + mo] = (half_t)val;
    }
  }
}

// ---------------------------------------------------------------------------
// conv3: implicit GEMM. M=64, N=49 (pad 64), K=576. 9x9x64 tile via TDM.
// ---------------------------------------------------------------------------
__global__ __launch_bounds__(256) void conv3_wmma(const half_t* act2, const half_t* W3p,
                                                  const float* b3, half_t* act3) {
  __shared__ half_t a2[9 * 9 * 64];  // 10.4 KB
  int img = blockIdx.x, tid = threadIdx.x, lane = tid & 31, wv = tid >> 5;
  stage_tile(a2, act2 + (size_t)img * 5184, 1296, tid);
  int hf = lane >> 4;
  for (int rr = 0; rr < 2; ++rr) {
    int tile = wv + rr * 8;  // 16 tiles = 4 m x 4 n
    int mt = tile & 3, nt = tile >> 2;
    int m = mt * 16 + (lane & 15);
    int npix = nt * 16 + (lane & 15);
    bool valid = npix < 49;
    int npc = valid ? npix : 48;
    int oy = npc / 7, ox = npc % 7;
    v8f acc = zero_v8f();
    for (int kk = 0; kk < 576; kk += 32) {
      ABu fa, fb;
#pragma unroll
      for (int v = 0; v < 8; ++v) {
        int ka = kk + ((v < 4) ? 0 : 16) + hf * 8 + (v & 3) * 2;
        fa.u[v] = *(const unsigned int*)(W3p + m * 576 + ka);
        int kb = kk + ((v < 4) ? 0 : 16) + hf * 8 + (v & 3) * 2;
        int p = kb >> 6, c = kb & 63;
        int ky = p / 3, kx = p % 3;
        int offb = ((oy + ky) * 9 + (ox + kx)) * 64 + c;
        fb.u[v] = *(const unsigned int*)(a2 + offb);
      }
      acc = __builtin_amdgcn_wmma_f32_16x16x32_f16(false, fa.v, false, fb.v,
                                                   (short)0, acc, false, false);
    }
#pragma unroll
    for (int v = 0; v < 8; ++v) {
      int mo = mt * 16 + (lane >> 4) * 8 + v;
      float val = fmaxf(acc[v] + b3[mo], 0.0f);
      if (valid) act3[((size_t)img * 49 + npix) * 64 + mo] = (half_t)val;
    }
  }
}

// ---------------------------------------------------------------------------
// Generic WMMA GEMM: Y[N][Mout] = X[N][K] * W[Mpad][K]^T (+bias, relu).
// Grid: (N/32, Mpad/64). 8 waves, each owns one 16x16 tile of the 32x64 block.
// Requires K % 64 == 0, N % 32 == 0.
// ---------------------------------------------------------------------------
__global__ __launch_bounds__(256) void gemm_xwt(const half_t* X, const half_t* W,
                                                const float* bias, half_t* Yh, float* Yf,
                                                int K, int Mout, int relu) {
  __shared__ half_t xs[32 * 64];   // 4 KB
  __shared__ half_t wsm[64 * 64];  // 8 KB
  int tid = threadIdx.x, lane = tid & 31, wv = tid >> 5;
  int n0 = blockIdx.x * 32, m0 = blockIdx.y * 64;
  int nt = wv & 1, mt = wv >> 1;
  v8f acc = zero_v8f();
  for (int kk = 0; kk < K; kk += 64) {
    {
      int row = tid >> 3, j = tid & 7;
      ((uint4*)xs)[tid] = *(const uint4*)(X + (size_t)(n0 + row) * K + kk + j * 8);
    }
#pragma unroll
    for (int q = 0; q < 2; ++q) {
      int i = tid + q * 256, row = i >> 3, j = i & 7;
      ((uint4*)wsm)[i] = *(const uint4*)(W + (size_t)(m0 + row) * K + kk + j * 8);
    }
    __syncthreads();
#pragma unroll
    for (int kh = 0; kh < 64; kh += 32) {
      v16h a = load_frag_a(wsm + mt * 16 * 64 + kh, 64, lane);
      v16h b = load_frag_b(xs + nt * 16 * 64 + kh, 64, lane);
      acc = __builtin_amdgcn_wmma_f32_16x16x32_f16(false, a, false, b, (short)0,
                                                   acc, false, false);
    }
    __syncthreads();
  }
  int n = n0 + nt * 16 + (lane & 15);
#pragma unroll
  for (int v = 0; v < 8; ++v) {
    int m = m0 + mt * 16 + (lane >> 4) * 8 + v;
    if (m < Mout) {
      float val = acc[v] + (bias ? bias[m] : 0.0f);
      if (relu) val = fmaxf(val, 0.0f);
      if (Yh) Yh[(size_t)n * Mout + m] = (half_t)val;
      if (Yf) Yf[(size_t)n * Mout + m] = val;
    }
  }
}

// ---------------------------------------------------------------------------
// LSTM step t, phase 1: build Xcat[64][1088] = [hidden(512) r(1) onehot(18) pad |
// masked h(512) pad] and apply keep-mask to c state in place.
// ---------------------------------------------------------------------------
__global__ __launch_bounds__(256) void lstm_prep(const float* r, const int* a,
                                                 const int* done, const half_t* hiddenh,
                                                 const half_t* hstate_h, float* cstate,
                                                 half_t* Xcat, int t) {
  int b = blockIdx.x, tid = threadIdx.x;
  float kf = 1.0f - (float)done[t * 64 + b];
  for (int h = tid; h < 512; h += 256) cstate[b * 512 + h] *= kf;
  const half_t* hid = hiddenh + (size_t)(t * 64 + b) * 512;
  half_t* xr = Xcat + b * 1088;
  float rv = r[t * 64 + b];
  int av = a[t * 64 + b];
  for (int c = tid; c < 1088; c += 256) {
    float val;
    if (c < 512) val = (float)hid[c];
    else if (c == 512) val = rv;
    else if (c < 531) val = (av == (c - 513)) ? 1.0f : 0.0f;
    else if (c < 544) val = 0.0f;
    else if (c < 1056) val = (float)hstate_h[b * 512 + (c - 544)] * kf;
    else val = 0.0f;
    xr[c] = (half_t)val;
  }
}

// ---------------------------------------------------------------------------
// LSTM step t, phase 2: gate GEMM (WMMA) + fused cell update.
// Grid (2, 8): 32 batches x 64-wide hidden slice per WG. Each WG computes the
// i/f/g/o rows of its slice (256 gate rows, K=1088) and updates h/c locally.
// ---------------------------------------------------------------------------
__global__ __launch_bounds__(256) void lstm_step(const half_t* Xcat, const half_t* WLp,
                                                 const float* bsum, float* cstate,
                                                 half_t* hstate_h, float* hstate_f,
                                                 half_t* hs, int t) {
  __shared__ __align__(16) char smem[36 * 1024];
  half_t* xs = (half_t*)smem;            // 32x64 = 4 KB
  half_t* wsm = (half_t*)(smem + 4096);  // 256x64 = 32 KB
  float* gbuf = (float*)smem;            // 32x256 f32 = 32 KB (reused after GEMM)

  int tid = threadIdx.x, lane = tid & 31, wv = tid >> 5;
  int n0 = blockIdx.x * 32, hs0 = blockIdx.y * 64;
  int nt = wv & 1, mtg = wv >> 1;  // n-tile, m-tile-within-gate
  v8f acc[4] = {zero_v8f(), zero_v8f(), zero_v8f(), zero_v8f()};

  for (int kk = 0; kk < 1088; kk += 64) {
    {
      int row = tid >> 3, j = tid & 7;
      ((uint4*)xs)[tid] = *(const uint4*)(Xcat + (size_t)(n0 + row) * 1088 + kk + j * 8);
    }
#pragma unroll
    for (int q = 0; q < 8; ++q) {
      int i = tid + q * 256, rr = i >> 3, j = i & 7;
      int grow = (rr >> 6) * 512 + hs0 + (rr & 63);  // gate-major global row
      ((uint4*)wsm)[i] = *(const uint4*)(WLp + (size_t)grow * 1088 + kk + j * 8);
    }
    __syncthreads();
#pragma unroll
    for (int g = 0; g < 4; ++g) {
      int mt = g * 4 + mtg;
#pragma unroll
      for (int kh = 0; kh < 64; kh += 32) {
        v16h a = load_frag_a(wsm + mt * 16 * 64 + kh, 64, lane);
        v16h b = load_frag_b(xs + nt * 16 * 64 + kh, 64, lane);
        acc[g] = __builtin_amdgcn_wmma_f32_16x16x32_f16(false, a, false, b, (short)0,
                                                        acc[g], false, false);
      }
    }
    __syncthreads();
  }

  // Spill accumulators to LDS gate buffer [32 n][256 m].
  int nloc = nt * 16 + (lane & 15);
#pragma unroll
  for (int g = 0; g < 4; ++g) {
    int mt = g * 4 + mtg;
#pragma unroll
    for (int v = 0; v < 8; ++v) {
      int mloc = mt * 16 + (lane >> 4) * 8 + v;
      gbuf[nloc * 256 + mloc] = acc[g][v];
    }
  }
  __syncthreads();

  // Fused cell update: 32 batches x 64 hidden = 2048 elems, 8 per thread.
#pragma unroll
  for (int q = 0; q < 8; ++q) {
    int e = tid + q * 256;
    int nl = e >> 6, hl = e & 63;
    int b = n0 + nl, h = hs0 + hl;
    float gi = gbuf[nl * 256 + hl] + bsum[hs0 + hl];
    float gf = gbuf[nl * 256 + 64 + hl] + bsum[512 + hs0 + hl];
    float gg = gbuf[nl * 256 + 128 + hl] + bsum[1024 + hs0 + hl];
    float go = gbuf[nl * 256 + 192 + hl] + bsum[1536 + hs0 + hl];
    float c_old = cstate[b * 512 + h];  // already keep-masked by lstm_prep
    float c_new = sigm(gf) * c_old + sigm(gi) * tanhf(gg);
    float h_new = sigm(go) * tanhf(c_new);
    cstate[b * 512 + h] = c_new;
    hstate_h[b * 512 + h] = (half_t)h_new;
    hstate_f[b * 512 + h] = h_new;
    hs[((size_t)(t * 64 + b)) * 512 + h] = (half_t)h_new;
  }
}

// ---------------------------------------------------------------------------
// Dueling combine: q = val + adv - mean(adv).
// ---------------------------------------------------------------------------
__global__ __launch_bounds__(256) void heads_combine(const float* adv, const float* val,
                                                     float* qout) {
  int n = blockIdx.x * 256 + threadIdx.x;
  if (n >= 5120) return;
  float s = 0.0f;
#pragma unroll
  for (int j = 0; j < 18; ++j) s += adv[n * 18 + j];
  float mean = s * (1.0f / 18.0f);
  float v = val[n];
#pragma unroll
  for (int j = 0; j < 18; ++j) qout[n * 18 + j] = v + adv[n * 18 + j] - mean;
}

__global__ __launch_bounds__(256) void copy_states(const float* hstate_f,
                                                   const float* cstate_f, float* out) {
  int i = blockIdx.x * 256 + threadIdx.x;
  if (i < 32768) {
    out[92160 + i] = hstate_f[i];
    out[92160 + 32768 + i] = cstate_f[i];
  }
}

// ---------------------------------------------------------------------------
extern "C" void kernel_launch(void* const* d_in, const int* in_sizes, int n_in,
                              void* d_out, int out_size, void* d_ws, size_t ws_size,
                              hipStream_t stream) {
  (void)in_sizes; (void)n_in; (void)out_size; (void)ws_size;
  const int*   o    = (const int*)d_in[0];
  const int*   a    = (const int*)d_in[1];
  const float* r    = (const float*)d_in[2];
  const int*   done = (const int*)d_in[3];
  const float* h0   = (const float*)d_in[4];
  const float* c0   = (const float*)d_in[5];
  const float* c1w  = (const float*)d_in[6];
  const float* c1b  = (const float*)d_in[7];
  const float* c2w  = (const float*)d_in[8];
  const float* c2b  = (const float*)d_in[9];
  const float* c3w  = (const float*)d_in[10];
  const float* c3b  = (const float*)d_in[11];
  const float* fcw  = (const float*)d_in[12];
  const float* fcb  = (const float*)d_in[13];
  const float* wih  = (const float*)d_in[14];
  const float* whh  = (const float*)d_in[15];
  const float* bih  = (const float*)d_in[16];
  const float* bhh  = (const float*)d_in[17];
  const float* aw1  = (const float*)d_in[18];
  const float* ab1  = (const float*)d_in[19];
  const float* aw2  = (const float*)d_in[20];
  const float* ab2  = (const float*)d_in[21];
  const float* vw1  = (const float*)d_in[22];
  const float* vb1  = (const float*)d_in[23];
  const float* vw2  = (const float*)d_in[24];
  const float* vb2  = (const float*)d_in[25];
  float* out = (float*)d_out;

  char* ws = (char*)d_ws;
  size_t off = 0;
  auto alloc = [&](size_t bytes) -> void* {
    void* p = ws + off;
    off += (bytes + 255) & ~(size_t)255;
    return p;
  };
  // Packed weights (f16) + combined bias.
  half_t* W1p = (half_t*)alloc(2048 * 2);
  half_t* W2p = (half_t*)alloc(32768 * 2);
  half_t* W3p = (half_t*)alloc(36864 * 2);
  half_t* FCp = (half_t*)alloc((size_t)512 * 3136 * 2);
  half_t* WLp = (half_t*)alloc((size_t)2048 * 1088 * 2);
  half_t* A1p = (half_t*)alloc((size_t)262144 * 2);
  half_t* A2p = (half_t*)alloc((size_t)32768 * 2);
  half_t* V1p = (half_t*)alloc((size_t)262144 * 2);
  half_t* V2p = (half_t*)alloc((size_t)32768 * 2);
  float*  bsum = (float*)alloc(2048 * 4);
  // Activations (f16 NHWC).
  half_t* act1    = (half_t*)alloc((size_t)5120 * 12800 * 2);  // 131 MB
  half_t* act2    = (half_t*)alloc((size_t)5120 * 5184 * 2);   // 53 MB
  half_t* act3    = (half_t*)alloc((size_t)5120 * 3136 * 2);   // 32 MB
  half_t* hiddenh = (half_t*)alloc((size_t)5120 * 512 * 2);
  half_t* hsbuf   = (half_t*)alloc((size_t)5120 * 512 * 2);
  half_t* advh    = (half_t*)alloc((size_t)5120 * 512 * 2);
  half_t* valh    = (half_t*)alloc((size_t)5120 * 512 * 2);
  half_t* Xcat    = (half_t*)alloc((size_t)64 * 1088 * 2);
  half_t* hstate_h = (half_t*)alloc((size_t)32768 * 2);
  float* adv2o    = (float*)alloc((size_t)5120 * 18 * 4);
  float* val2o    = (float*)alloc((size_t)5120 * 4);
  float* hstate_f = (float*)alloc((size_t)32768 * 4);
  float* cstate_f = (float*)alloc((size_t)32768 * 4);

  pack_weights<<<2048, 256, 0, stream>>>(c1w, c2w, c3w, fcw, wih, whh, bih, bhh,
                                         aw1, aw2, vw1, vw2, W1p, W2p, W3p, FCp,
                                         WLp, A1p, A2p, V1p, V2p, bsum);
  init_states<<<128, 256, 0, stream>>>(h0, c0, hstate_h, hstate_f, cstate_f);

  // CNN trunk.
  conv1_direct<<<5120, 256, 0, stream>>>(o, c1w, c1b, act1);
  conv2_wmma<<<5120, 256, 0, stream>>>(act1, W2p, c2b, act2);
  conv3_wmma<<<5120, 256, 0, stream>>>(act2, W3p, c3b, act3);
  gemm_xwt<<<dim3(160, 8), 256, 0, stream>>>(act3, FCp, fcb, hiddenh, nullptr,
                                             3136, 512, 1);

  // LSTM scan: stream ordering provides per-step synchronization.
  for (int t = 0; t < 80; ++t) {
    lstm_prep<<<64, 256, 0, stream>>>(r, a, done, hiddenh, hstate_h, cstate_f, Xcat, t);
    lstm_step<<<dim3(2, 8), 256, 0, stream>>>(Xcat, WLp, bsum, cstate_f, hstate_h,
                                              hstate_f, hsbuf, t);
  }

  // Dueling heads.
  gemm_xwt<<<dim3(160, 8), 256, 0, stream>>>(hsbuf, A1p, ab1, advh, nullptr, 512, 512, 1);
  gemm_xwt<<<dim3(160, 1), 256, 0, stream>>>(advh, A2p, ab2, nullptr, adv2o, 512, 18, 0);
  gemm_xwt<<<dim3(160, 8), 256, 0, stream>>>(hsbuf, V1p, vb1, valh, nullptr, 512, 512, 1);
  gemm_xwt<<<dim3(160, 1), 256, 0, stream>>>(valh, V2p, vb2, nullptr, val2o, 512, 1, 0);

  heads_combine<<<20, 256, 0, stream>>>(adv2o, val2o, out);
  copy_states<<<128, 256, 0, stream>>>(hstate_f, cstate_f, out);
  (void)W1p;  // conv1 uses f32 weights directly; W1p kept for layout symmetry.
}
